// FastAttention_11682311045351
// MI455X (gfx1250) — matrix operand order repaired and checked by reading, not compiled, not run on hardware
//
#include <hip/hip_runtime.h>
#include <stdint.h>

typedef _Float16 half_t;
typedef __attribute__((ext_vector_type(16))) _Float16     v16h;
typedef __attribute__((ext_vector_type(8)))  float        v8f;
typedef __attribute__((ext_vector_type(8)))  unsigned int v8u;

#define N_CTX   8192
#define D_HEAD  64
#define M_REAL  266
#define M_PAD   288          // 18 tiles of 16 ; 9 K-chunks of 32
#define BH      32

__device__ __constant__ const float DN    = 0.35355339059327379f;   // 64^-0.25
__device__ __constant__ const float RATIO = 0.061313933948496583f;  // 266^-0.5
__device__ __constant__ const float EPSV  = 1e-4f;

// ---- workspace layout (in floats) ----
//  [0,32)                      : per-head encoded max (uint32, monotone float encoding)
//  [32, 32+BH*M_PAD)           : ksum  f32
//  [.. , ..+BH*M_PAD*64)       : ctx   f32
//  [PROJ_OFF ...)              : projection, f16, padded [M_PAD][64]
#define KSUM_OFF 32
#define CTX_OFF  (32 + BH * M_PAD)
#define PROJ_OFF (CTX_OFF + BH * M_PAD * 64)   // = 599072 floats
#define WS_ZTOT  PROJ_OFF

// ---------------- monotone float<->uint encoding for atomic max ----------------
__device__ __forceinline__ unsigned encF(float f) {
  unsigned u = __float_as_uint(f);
  return (u & 0x80000000u) ? ~u : (u | 0x80000000u);
}
__device__ __forceinline__ float decF(unsigned u) {
  return __uint_as_float((u & 0x80000000u) ? (u & 0x7FFFFFFFu) : ~u);
}

// ---------------- WMMA helpers ----------------
__device__ __forceinline__ v8f wmma_f16(v16h a, v16h b, v8f c) {
  return __builtin_amdgcn_wmma_f32_16x16x32_f16(false, a, false, b, (short)0, c,
                                                false, false);
}

// A fragment (16x32 f16) from [row][k]-contiguous half array.
// lane: row = lane&15, k-offset = 8*(lane>>4); element pairs are consecutive k.
__device__ __forceinline__ v16h fragA(const half_t* base, int stride, int lane) {
  const int row = lane & 15;
  const int ko  = (lane >> 4) << 3;
  const half_t* p = base + row * stride;
  union { v16h h; v8u u; } f;
#pragma unroll
  for (int v = 0; v < 8; ++v) {
    const int k2 = (v < 4) ? (ko + 2 * v) : (16 + ko + 2 * (v - 4));
    f.u[v] = *(const uint32_t*)(p + k2);
  }
  return f.h;
}

// B fragment (32x16 f16) from [col][k]-contiguous half array.
// lane: col = lane&15, k-offset = 8*(lane>>4); same k order as fragA.
__device__ __forceinline__ v16h fragB(const half_t* base, int stride, int lane) {
  const int col = lane & 15;
  const int kh  = (lane >> 4) << 3;
  const half_t* p = base + col * stride;
  union { v16h h; v8u u; } f;
#pragma unroll
  for (int v = 0; v < 8; ++v) f.u[v] = *(const uint32_t*)(p + ((v < 4) ? (kh + 2 * v) : (16 + kh + 2 * (v - 4))));
  return f.h;
}

// Stage a 16x64 f32 tile -> f16 LDS [16][64], scaled; diag16[row] = 0.5*sum((s*x)^2).
__device__ __forceinline__ void stage16x64(const float* src, half_t* dst,
                                           float* diag16, int lane, float scale) {
  const int row = lane >> 1, c0 = (lane & 1) * 32;
  const float4* s = (const float4*)(src + row * 64 + c0);
  half_t* d = dst + row * 64 + c0;
  float ss = 0.f;
#pragma unroll
  for (int j = 0; j < 8; ++j) {
    float4 f = s[j];
    float a0 = f.x * scale, a1 = f.y * scale, a2 = f.z * scale, a3 = f.w * scale;
    ss += a0 * a0 + a1 * a1 + a2 * a2 + a3 * a3;
    d[4 * j + 0] = (half_t)a0; d[4 * j + 1] = (half_t)a1;
    d[4 * j + 2] = (half_t)a2; d[4 * j + 3] = (half_t)a3;
  }
  ss += __shfl_xor(ss, 1, 32);
  if ((lane & 1) == 0) diag16[row] = 0.5f * ss;
}

// =================== K0: init workspace ===================
__global__ __launch_bounds__(256) void perf_k0_init(const float* __restrict__ proj,
                                                    float* __restrict__ ws) {
  const int i = blockIdx.x * 256 + threadIdx.x;
  if (i < 32) {
    ((unsigned*)ws)[i] = 0x007FFFFFu;                 // encF(-inf)
  } else if (i < WS_ZTOT) {
    ws[i] = 0.f;
  }
  if (i < M_PAD * 64) {
    half_t* ph = (half_t*)(ws + PROJ_OFF);
    const int m = i >> 6, d = i & 63;
    ph[i] = (half_t)((m < M_REAL) ? proj[m * 64 + d] : 0.f);
  }
}

// =================== K1: global max of k's data_dash (per head) ===================
__global__ __launch_bounds__(256) void perf_k1_kmax(const float* __restrict__ kdat,
                                                    float* __restrict__ ws) {
  const int head = blockIdx.y;
  const int tile = blockIdx.x;              // 64 tiles of 128 rows
  const int tid = threadIdx.x, lane = tid & 31, wave = tid >> 5;

  __shared__ half_t s_proj[M_PAD * 64];
  __shared__ half_t s_stage[8][16 * 64];
  __shared__ float  s_diag[8][16];
  __shared__ float  s_wmax[8];

  const half_t* projH = (const half_t*)(ws + PROJ_OFF);
  for (int i = tid; i < M_PAD * 64 / 2; i += 256)
    ((uint32_t*)s_proj)[i] = ((const uint32_t*)projH)[i];
  __syncthreads();

  const float* src = kdat + ((size_t)head * N_CTX + tile * 128 + wave * 16) * 64;
  stage16x64(src, &s_stage[wave][0], &s_diag[wave][0], lane, DN);

  const v16h a0 = fragA(&s_stage[wave][0], 64, lane);
  const v16h a1 = fragA(&s_stage[wave][32], 64, lane);

  float mx = -3.0e38f;
  for (int mt = 0; mt < M_PAD / 16; ++mt) {
    const v16h b0 = fragB(&s_proj[(mt * 16) * 64 + 0], 64, lane);
    const v16h b1 = fragB(&s_proj[(mt * 16) * 64 + 32], 64, lane);
    v8f c = {0.f, 0.f, 0.f, 0.f, 0.f, 0.f, 0.f, 0.f};
    c = wmma_f16(a0, b0, c);
    c = wmma_f16(a1, b1, c);
    const int m = mt * 16 + (lane & 15);
    if (m < M_REAL) {
#pragma unroll
      for (int r = 0; r < 8; ++r) mx = fmaxf(mx, c[r]);
    }
  }
#pragma unroll
  for (int off = 16; off >= 1; off >>= 1) mx = fmaxf(mx, __shfl_xor(mx, off, 32));
  if (lane == 0) s_wmax[wave] = mx;
  __syncthreads();
  if (tid == 0) {
    float bm = s_wmax[0];
#pragma unroll
    for (int w = 1; w < 8; ++w) bm = fmaxf(bm, s_wmax[w]);
    atomicMax((unsigned*)ws + head, encF(bm));
  }
}

// =================== K2: kp features, ctx = kp^T * v, ksum ===================
__global__ __launch_bounds__(256) void perf_k2_context(const float* __restrict__ kdat,
                                                       const float* __restrict__ vdat,
                                                       float* __restrict__ ws) {
  const int head = blockIdx.y;
  const int chunk = blockIdx.x;             // 8 chunks of 1024 rows
  const int tid = threadIdx.x, lane = tid & 31, wave = tid >> 5;

  float* ksumG = ws + KSUM_OFF;
  float* ctxG  = ws + CTX_OFF;
  const half_t* projH = (const half_t*)(ws + PROJ_OFF);

  __shared__ half_t s_proj[M_PAD * 64];          // 36864 B
  __shared__ half_t s_stage[8][16 * 64];         // 16384 B
  __shared__ half_t s_kph[M_PAD * 128];          // kp transposed [m][n]  73728 B
  __shared__ half_t s_vT[64 * 128];              // v transposed  [e][n]  16384 B
  __shared__ float  s_ksum[M_PAD];
  __shared__ float  s_diag[8][16];
  __shared__ float  s_gmax;

  for (int i = tid; i < M_PAD * 64 / 2; i += 256)
    ((uint32_t*)s_proj)[i] = ((const uint32_t*)projH)[i];
  for (int i = tid; i < M_PAD; i += 256) s_ksum[i] = 0.f;
  if (tid == 0) s_gmax = decF(((const unsigned*)ws)[head]);
  __syncthreads();
  const float gmax = s_gmax;

  v8f acc[9];
  const v8f vzero = {0.f, 0.f, 0.f, 0.f, 0.f, 0.f, 0.f, 0.f};
#pragma unroll
  for (int i = 0; i < 9; ++i) acc[i] = vzero;

  const float* kbase = kdat + ((size_t)head * N_CTX + chunk * 1024) * 64;
  const float* vbase = vdat + ((size_t)head * N_CTX + chunk * 1024) * 64;

  for (int slab = 0; slab < 8; ++slab) {      // 8 * 128 = 1024 rows
    // (a) per-wave: stage k tile, GEMM1 -> kp, write kp^T to LDS, ksum atomics
    stage16x64(kbase + (slab * 128 + wave * 16) * 64, &s_stage[wave][0],
               &s_diag[wave][0], lane, DN);
    const v16h a0 = fragA(&s_stage[wave][0], 64, lane);
    const v16h a1 = fragA(&s_stage[wave][32], 64, lane);
    float diag[8];
    const int halfg = lane >> 4;
#pragma unroll
    for (int r = 0; r < 8; ++r) diag[r] = s_diag[wave][8 * halfg + r];

    for (int mt = 0; mt < M_PAD / 16; ++mt) {
      const v16h b0 = fragB(&s_proj[(mt * 16) * 64 + 0], 64, lane);
      const v16h b1 = fragB(&s_proj[(mt * 16) * 64 + 32], 64, lane);
      v8f c = vzero;
      c = wmma_f16(a0, b0, c);
      c = wmma_f16(a1, b1, c);
      const int m = mt * 16 + (lane & 15);
      const bool valid = m < M_REAL;
      float ksacc = 0.f;
#pragma unroll
      for (int r = 0; r < 8; ++r) {
        const float kp =
            valid ? (RATIO * (__expf(c[r] - diag[r] - gmax) + EPSV)) : 0.f;
        ksacc += kp;
        s_kph[m * 128 + wave * 16 + 8 * halfg + r] = (half_t)kp;
      }
      if (valid) atomicAdd(&s_ksum[m], ksacc);
    }

    // (b) cooperative: stage v slab transposed [e][n]
    {
      const float* vt = vbase + slab * 128 * 64;
      const int row = tid >> 1, c0 = (tid & 1) * 32;
      const float4* src = (const float4*)(vt + row * 64 + c0);
#pragma unroll
      for (int j = 0; j < 8; ++j) {
        float4 f = src[j];
        s_vT[(c0 + 4 * j + 0) * 128 + row] = (half_t)f.x;
        s_vT[(c0 + 4 * j + 1) * 128 + row] = (half_t)f.y;
        s_vT[(c0 + 4 * j + 2) * 128 + row] = (half_t)f.z;
        s_vT[(c0 + 4 * j + 3) * 128 + row] = (half_t)f.w;
      }
    }
    __syncthreads();

    // (d) GEMM2: ctx[mt][et] += kp^T * v over this 128-row slab (K = 128)
#pragma unroll
    for (int i = 0; i < 9; ++i) {
      const int t = wave + 8 * i;             // 72 tiles, 9 per wave
      const int mt = t >> 2, et = t & 3;
      v8f c = acc[i];
#pragma unroll
      for (int kc = 0; kc < 4; ++kc) {
        const v16h a = fragA(&s_kph[(mt * 16) * 128 + kc * 32], 128, lane);
        const v16h b = fragB(&s_vT[(et * 16) * 128 + kc * 32], 128, lane);
        c = wmma_f16(a, b, c);
      }
      acc[i] = c;
    }
    __syncthreads();
  }

  // flush partial ctx / ksum to workspace
#pragma unroll
  for (int i = 0; i < 9; ++i) {
    const int t = wave + 8 * i;
    const int mt = t >> 2, et = t & 3;
    const int mrow = mt * 16 + 8 * (lane >> 4);
    const int e = et * 16 + (lane & 15);
#pragma unroll
    for (int r = 0; r < 8; ++r)
      atomicAdd(&ctxG[((size_t)head * M_PAD + mrow + r) * 64 + e], acc[i][r]);
  }
  for (int i = tid; i < M_PAD; i += 256)
    atomicAdd(&ksumG[head * M_PAD + i], s_ksum[i]);
}

// =================== K3: qp features, out = (qp*ctx) / (qp*ksum) ===================
__global__ __launch_bounds__(256) void perf_k3_out(const float* __restrict__ qdat,
                                                   float* __restrict__ out,
                                                   float* __restrict__ ws) {
  const int head = blockIdx.y;
  const int tile = blockIdx.x;              // 64 tiles of 128 rows
  const int tid = threadIdx.x, lane = tid & 31, wave = tid >> 5;

  const float* ksumG = ws + KSUM_OFF;
  const float* ctxG  = ws + CTX_OFF;
  const half_t* projH = (const half_t*)(ws + PROJ_OFF);

  __shared__ half_t s_proj[M_PAD * 64];          // 36864 B
  __shared__ half_t s_ctxT[64 * M_PAD];          // ctx transposed [e][m] f16, 36864 B
  __shared__ float  s_ksum[M_PAD];
  __shared__ half_t s_wbuf[8][16 * M_PAD];       // per-wave: q stage then qp, 73728 B
  __shared__ float  s_diag[8][16];

  for (int i = tid; i < M_PAD * 64 / 2; i += 256)
    ((uint32_t*)s_proj)[i] = ((const uint32_t*)projH)[i];
  {
    const float* cg = ctxG + (size_t)head * M_PAD * 64;
    for (int i = tid; i < M_PAD * 64; i += 256) {
      const int m = i >> 6, e = i & 63;
      s_ctxT[e * M_PAD + m] = (half_t)cg[i];
    }
  }
  for (int i = tid; i < M_PAD; i += 256) s_ksum[i] = ksumG[head * M_PAD + i];
  __syncthreads();

  // per-wave: stage q tile (reuses front of s_wbuf)
  const float* src = qdat + ((size_t)head * N_CTX + tile * 128 + wave * 16) * 64;
  stage16x64(src, &s_wbuf[wave][0], &s_diag[wave][0], lane, DN);
  const v16h a0 = fragA(&s_wbuf[wave][0], 64, lane);
  const v16h a1 = fragA(&s_wbuf[wave][32], 64, lane);

  const v8f vzero = {0.f, 0.f, 0.f, 0.f, 0.f, 0.f, 0.f, 0.f};
  v8f ddv[M_PAD / 16];
#pragma unroll
  for (int mt = 0; mt < M_PAD / 16; ++mt) {
    const v16h b0 = fragB(&s_proj[(mt * 16) * 64 + 0], 64, lane);
    const v16h b1 = fragB(&s_proj[(mt * 16) * 64 + 32], 64, lane);
    v8f c = vzero;
    c = wmma_f16(a0, b0, c);
    c = wmma_f16(a1, b1, c);
    ddv[mt] = c;
  }

  // per-row max over valid features
  const int halfg = lane >> 4;
  float rmax[8];
#pragma unroll
  for (int r = 0; r < 8; ++r) rmax[r] = -3.0e38f;
#pragma unroll
  for (int mt = 0; mt < M_PAD / 16; ++mt) {
    const int m = mt * 16 + (lane & 15);
    if (m < M_REAL) {
#pragma unroll
      for (int r = 0; r < 8; ++r) rmax[r] = fmaxf(rmax[r], ddv[mt][r]);
    }
  }
#pragma unroll
  for (int off = 1; off <= 8; off <<= 1) {
#pragma unroll
    for (int r = 0; r < 8; ++r) rmax[r] = fmaxf(rmax[r], __shfl_xor(rmax[r], off, 32));
  }

  float diag[8];
#pragma unroll
  for (int r = 0; r < 8; ++r) diag[r] = s_diag[wave][8 * halfg + r];

  // qp = ratio*(exp(dd - diag - rowmax)+eps); write to LDS; accumulate qp.ksum
  // (padded m>=266 need no mask: ksum and ctx rows there are exactly zero)
  half_t* qph = &s_wbuf[wave][0];
  float dsum[8] = {0.f, 0.f, 0.f, 0.f, 0.f, 0.f, 0.f, 0.f};
#pragma unroll
  for (int mt = 0; mt < M_PAD / 16; ++mt) {
    const int mcol = mt * 16 + (lane & 15);
    const float ks = s_ksum[mcol];
#pragma unroll
    for (int r = 0; r < 8; ++r) {
      const float qp = RATIO * (__expf(ddv[mt][r] - diag[r] - rmax[r]) + EPSV);
      dsum[r] += qp * ks;
      qph[(8 * halfg + r) * M_PAD + mcol] = (half_t)qp;
    }
  }
#pragma unroll
  for (int off = 1; off <= 8; off <<= 1) {
#pragma unroll
    for (int r = 0; r < 8; ++r) dsum[r] += __shfl_xor(dsum[r], off, 32);
  }
  float dinv[8];
#pragma unroll
  for (int r = 0; r < 8; ++r) dinv[r] = 1.f / dsum[r];

  // GEMM2: out[16][64] = qp[16][288] * ctx[288][64]
  const size_t nbase = (size_t)head * N_CTX + tile * 128 + wave * 16 + 8 * halfg;
#pragma unroll
  for (int et = 0; et < 4; ++et) {
    v8f c = vzero;
#pragma unroll
    for (int kc = 0; kc < M_PAD / 32; ++kc) {
      const v16h a = fragA(qph + kc * 32, M_PAD, lane);
      const v16h b = fragB(&s_ctxT[(et * 16) * M_PAD + kc * 32], M_PAD, lane);
      c = wmma_f16(a, b, c);
    }
    const int e = et * 16 + (lane & 15);
#pragma unroll
    for (int r = 0; r < 8; ++r) out[(nbase + r) * 64 + e] = c[r] * dinv[r];
  }
}

// =================== launch ===================
extern "C" void kernel_launch(void* const* d_in, const int* in_sizes, int n_in,
                              void* d_out, int out_size, void* d_ws, size_t ws_size,
                              hipStream_t stream) {
  const float* q    = (const float*)d_in[0];
  const float* k    = (const float*)d_in[1];
  const float* v    = (const float*)d_in[2];
  const float* proj = (const float*)d_in[3];
  float* out = (float*)d_out;
  float* ws  = (float*)d_ws;

  const int initThreads = WS_ZTOT;  // also covers M_PAD*64 proj conversion
  perf_k0_init<<<(initThreads + 255) / 256, 256, 0, stream>>>(proj, ws);

  dim3 g1(N_CTX / 128, BH);
  perf_k1_kmax<<<g1, 256, 0, stream>>>(k, ws);

  dim3 g2(8, BH);
  perf_k2_context<<<g2, 256, 0, stream>>>(k, v, ws);

  dim3 g3(N_CTX / 128, BH);
  perf_k3_out<<<g3, 256, 0, stream>>>(q, out, ws);
}
